// StyledConv_1666447310864
// MI455X (gfx1250) — matrix operand order repaired and checked by reading, not compiled, not run on hardware
//
#include <hip/hip_runtime.h>

#define BB 4
#define IC 32
#define OC 32
#define HH 512
#define WW 512
#define SD 512

#define TPAD 34     // padded ic dimension in ushorts (17-dword stride -> conflict-free)
#define XCOLS 68    // padded tile width (66 used: 64 outputs + 2 halo)
#define XROWS 6     // 4 output rows + 2 halo

typedef __attribute__((ext_vector_type(16))) __bf16 v16bf;
typedef __attribute__((ext_vector_type(8)))  float  v8f;

__device__ __forceinline__ unsigned short f2bf(float f) {
  unsigned u = __builtin_bit_cast(unsigned, f);
  u += 0x7FFFu + ((u >> 16) & 1u);          // round-to-nearest-even
  return (unsigned short)(u >> 16);
}

// ---------------------------------------------------------------------------
// Prep: style GEMM (4x512x32), demod, and fold style*demod*wscale into bf16
// weights: wmod[b][tap][oc][ic]  (4*9*32*32 bf16 = 72KB in d_ws)
// ---------------------------------------------------------------------------
__global__ void styled_prep(const float* __restrict__ style_in,
                            const float* __restrict__ W_mod,
                            const float* __restrict__ b_mod,
                            const float* __restrict__ weight,
                            const float* __restrict__ style_inv,
                            unsigned short* __restrict__ wmod) {
  __shared__ float s_style[BB * IC];
  __shared__ float s_demod[OC];
  const float mod_scale = 0.04419417382415922f;   // 1/sqrt(512)
  const float w_scale   = 0.05892556509887896f;   // 1/sqrt(32*9)
  const int t = threadIdx.x;

  if (t < BB * IC) {                // style[b][c] = style_in @ (W_mod*s)^T + b_mod
    int b = t >> 5, c = t & 31;
    const float* si = style_in + b * SD;
    const float* wm = W_mod + c * SD;
    float acc = 0.f;
    for (int s = 0; s < SD; ++s) acc += si[s] * wm[s];
    s_style[t] = acc * mod_scale + b_mod[c];
  } else if (t < BB * IC + OC) {    // demod[oc] = rsqrt(sum((w*style_inv)^2)+eps)
    int oc = t - BB * IC;
    float acc = 0.f;
    for (int ic = 0; ic < IC; ++ic) {
      float sv = style_inv[ic];
      for (int k = 0; k < 9; ++k) {
        float w = weight[(oc * IC + ic) * 9 + k] * w_scale * sv;
        acc += w * w;
      }
    }
    s_demod[oc] = rsqrtf(acc + 1e-8f);
  }
  __syncthreads();

  for (int idx = t; idx < BB * 9 * OC * IC; idx += blockDim.x) {
    int b    = idx / (9 * OC * IC);
    int rem  = idx % (9 * OC * IC);
    int tap  = rem / (OC * IC);
    int rem2 = rem % (OC * IC);
    int oc   = rem2 >> 5;
    int ic   = rem2 & 31;
    float v = weight[(oc * IC + ic) * 9 + tap] * w_scale *
              s_style[b * IC + ic] * s_demod[oc];
    wmod[idx] = f2bf(v);
  }
}

// ---------------------------------------------------------------------------
// Conv as implicit GEMM with v_wmma_f32_16x16x32_bf16.
// Block: 256 thr (8 waves) -> one (b, 4-row, 64-col) tile of all 32 OC.
// Wave (r,h): 32 oc x 32 px => 4 accumulators; 9 taps x 4 WMMA = 36 WMMA.
// ---------------------------------------------------------------------------
__global__ __launch_bounds__(256)
void styled_conv(const float* __restrict__ x,
                 const float* __restrict__ noise,
                 const unsigned short* __restrict__ wmod,
                 const float* __restrict__ noise_w,
                 const float* __restrict__ act_bias,
                 float* __restrict__ out) {
  __shared__ __align__(16) unsigned short lds_x[XROWS * XCOLS * TPAD]; // [row][col][ic]
  __shared__ __align__(16) unsigned short lds_w[9 * OC * TPAD];        // [tap][oc][ic]

  const int b   = blockIdx.z;
  const int y0  = blockIdx.y * 4;
  const int x0  = blockIdx.x * 64;
  const int tid = threadIdx.x;

  // ---- stage x halo tile, f32 -> bf16, transposed to [row][col][ic]
  for (int i = tid; i < XROWS * 66 * IC; i += 256) {
    int rr  = i / (66 * IC);
    int rem = i % (66 * IC);
    int ic  = rem / 66;
    int cc  = rem % 66;
    int yy  = y0 - 1 + rr;
    int xx  = x0 - 1 + cc;
    float v = 0.f;
    if ((unsigned)yy < HH && (unsigned)xx < WW)
      v = x[((b * IC + ic) * HH + yy) * WW + xx];
    lds_x[(rr * XCOLS + cc) * TPAD + ic] = f2bf(v);
  }
  // ---- stage folded bf16 weights for this batch
  for (int i = tid; i < 9 * OC * IC; i += 256) {
    int tap = i / (OC * IC);
    int rem = i % (OC * IC);
    int oc  = rem >> 5;
    int ic  = rem & 31;
    lds_w[(tap * OC + oc) * TPAD + ic] = wmod[((b * 9 + tap) * OC + oc) * IC + ic];
  }
  __syncthreads();

  const int wave = tid >> 5;
  const int lane = tid & 31;
  const int r  = wave >> 1;   // output row in tile (0..3)
  const int h  = wave & 1;    // 32-col half (0..1)
  const int mn = lane & 15;   // M (A) / N (B,C) index
  const int hi = lane >> 4;

  union Frag { unsigned u[8]; v16bf v; };

  v8f acc00 = {}, acc01 = {}, acc10 = {}, acc11 = {};

  for (int tap = 0; tap < 9; ++tap) {
    const int dy = tap / 3 - 1, dx = tap % 3 - 1;

    // A fragments (16x32 bf16): lanes 0-15 hold K 0-15, lanes 16-31 K 16-31
    Frag a0, a1;
    {
      const int b1 = hi * 8;
      const unsigned* w0 = (const unsigned*)&lds_w[(tap * OC + mn) * TPAD];
      const unsigned* w1 = (const unsigned*)&lds_w[(tap * OC + 16 + mn) * TPAD];
      const int o0 = b1;
      const int o1 = b1 + 4;
      #pragma unroll
      for (int j = 0; j < 4; ++j) {
        a0.u[j]     = w0[o0 + j];
        a0.u[4 + j] = w0[o1 + j];
        a1.u[j]     = w1[o0 + j];
        a1.u[4 + j] = w1[o1 + j];
      }
    }

    const int trow = r + 1 + dy;
    #pragma unroll
    for (int nt = 0; nt < 2; ++nt) {
      // B fragment (32x16 bf16): lanes 0-15 K=0-15, lanes 16-31 K=16-31
      const int tcol = h * 32 + nt * 16 + mn + 1 + dx;
      const unsigned* xp =
          (const unsigned*)&lds_x[(trow * XCOLS + tcol) * TPAD + hi * 16];
      Frag bf;
      #pragma unroll
      for (int j = 0; j < 8; ++j) bf.u[j] = xp[j];

      if (nt == 0) {
        acc00 = __builtin_amdgcn_wmma_f32_16x16x32_bf16(false, a0.v, false, bf.v,
                                                        (short)0, acc00, false, false);
        acc10 = __builtin_amdgcn_wmma_f32_16x16x32_bf16(false, a1.v, false, bf.v,
                                                        (short)0, acc10, false, false);
      } else {
        acc01 = __builtin_amdgcn_wmma_f32_16x16x32_bf16(false, a0.v, false, bf.v,
                                                        (short)0, acc01, false, false);
        acc11 = __builtin_amdgcn_wmma_f32_16x16x32_bf16(false, a1.v, false, bf.v,
                                                        (short)0, acc11, false, false);
      }
    }
  }

  // ---- epilogue: + noise_w*noise + act_bias, leaky(0.2) * sqrt(2)
  const float nw = noise_w[0];
  const int   y  = y0 + r;
  const float SQRT2 = 1.4142135623730951f;
  #pragma unroll
  for (int nt = 0; nt < 2; ++nt) {
    const int xc = x0 + h * 32 + nt * 16 + mn;
    const float nz = nw * noise[(b * HH + y) * WW + xc];
    const v8f accA = nt ? acc01 : acc00;   // oc tile 0
    const v8f accB = nt ? acc11 : acc10;   // oc tile 1
    #pragma unroll
    for (int j = 0; j < 8; ++j) {
      {
        const int oc = j + 8 * hi;
        float v = accA[j] + nz + act_bias[oc];
        v = (v > 0.f ? v : 0.2f * v) * SQRT2;
        out[((b * OC + oc) * HH + y) * WW + xc] = v;
      }
      {
        const int oc = 16 + j + 8 * hi;
        float v = accB[j] + nz + act_bias[oc];
        v = (v > 0.f ? v : 0.2f * v) * SQRT2;
        out[((b * OC + oc) * HH + y) * WW + xc] = v;
      }
    }
  }
}

// ---------------------------------------------------------------------------
extern "C" void kernel_launch(void* const* d_in, const int* in_sizes, int n_in,
                              void* d_out, int out_size, void* d_ws, size_t ws_size,
                              hipStream_t stream) {
  const float* x         = (const float*)d_in[0];
  const float* style_in  = (const float*)d_in[1];
  const float* noise     = (const float*)d_in[2];
  const float* W_mod     = (const float*)d_in[3];
  const float* b_mod     = (const float*)d_in[4];
  const float* weight    = (const float*)d_in[5];
  const float* style_inv = (const float*)d_in[6];
  const float* noise_w   = (const float*)d_in[7];
  const float* act_bias  = (const float*)d_in[8];

  unsigned short* wmod = (unsigned short*)d_ws;   // 4*9*32*32 bf16 = 73728 B

  styled_prep<<<1, 256, 0, stream>>>(style_in, W_mod, b_mod, weight, style_inv, wmod);

  dim3 grid(WW / 64, HH / 4, BB);
  styled_conv<<<grid, 256, 0, stream>>>(x, noise, wmod, noise_w, act_bias,
                                        (float*)d_out);
}
